// AttentionLayer_55293408969192
// MI455X (gfx1250) — matrix operand order repaired and checked by reading, not compiled, not run on hardware
//
#include <hip/hip_runtime.h>
#include <hip/hip_bf16.h>
#include <math.h>

typedef __attribute__((ext_vector_type(16))) _Float16 v16h;
typedef __attribute__((ext_vector_type(8)))  _Float16 v8h;
typedef __attribute__((ext_vector_type(8)))  float    v8f;

#define CB 4
#define CS 1024
#define CD 1024
#define CH 16
#define CDH 64
#define NEGMASK (-100000000000.0f)

// ---------------------------------------------------------------------------
// helpers
// ---------------------------------------------------------------------------
__device__ __forceinline__ v8f wmma_f16(v16h a, v16h b, v8f c) {
  return __builtin_amdgcn_wmma_f32_16x16x32_f16(false, a, false, b, (short)0, c,
                                                false, false);
}

__device__ __forceinline__ v16h cat8(v8h lo, v8h hi) {
  return __builtin_shufflevector(lo, hi, 0, 1, 2, 3, 4, 5, 6, 7, 8, 9, 10, 11,
                                 12, 13, 14, 15);
}

// A-layout fragment: lane row fixed, K octet-interleaved:
//   e 0..7  -> k = g*8 + (e&7)
//   e 8..15 -> k = 16 + g*8 + (e&7)
// p = row base of this lane's 32-K chunk (global or LDS), g = lane>>4
template <typename P>
__device__ __forceinline__ v16h load_afrag(P p, int g) {
  v8h lo = *(const v8h*)(p + g * 8);
  v8h hi = *(const v8h*)(p + 16 + g * 8);
  return cat8(lo, hi);
}

// B-layout fragment: lane col fixed, K octet-interleaved (same as A)
// p = column base (K-contiguous storage), g = lane>>4
__device__ __forceinline__ v16h load_bfrag(const _Float16* p, int g) {
  v8h lo = *(const v8h*)(p + g * 8);
  v8h hi = *(const v8h*)(p + 16 + g * 8);
  return cat8(lo, hi);
}

__device__ __forceinline__ v8h cvt8(const float* p) {
  v8h r;
#pragma unroll
  for (int i = 0; i < 8; ++i) r[i] = (_Float16)p[i];
  return r;
}

__device__ __forceinline__ float rowmax16(float v) {
  v = fmaxf(v, __shfl_xor(v, 1, 32));
  v = fmaxf(v, __shfl_xor(v, 2, 32));
  v = fmaxf(v, __shfl_xor(v, 4, 32));
  v = fmaxf(v, __shfl_xor(v, 8, 32));
  return v;
}
__device__ __forceinline__ float rowsum16(float v) {
  v += __shfl_xor(v, 1, 32);
  v += __shfl_xor(v, 2, 32);
  v += __shfl_xor(v, 4, 32);
  v += __shfl_xor(v, 8, 32);
  return v;
}

// ---------------------------------------------------------------------------
// stage 0: conversions
// ---------------------------------------------------------------------------
__global__ void cvt_f32_to_f16(const float* __restrict__ in,
                               _Float16* __restrict__ out, int n) {
  int i = blockIdx.x * blockDim.x + threadIdx.x;
  if (i < n) out[i] = (_Float16)in[i];
}

// in: batch x rows x cols (f32) -> out: batch x cols x rows (f16)
__global__ void cvt_transpose_f16(const float* __restrict__ in,
                                  _Float16* __restrict__ out, int batch,
                                  int rows, int cols) {
  int i = blockIdx.x * blockDim.x + threadIdx.x;
  int per = rows * cols;
  if (i < batch * per) {
    int b = i / per;
    int rem = i - b * per;
    int r = rem / cols;
    int c = rem - r * cols;
    out[(size_t)b * per + (size_t)c * rows + r] = (_Float16)in[i];
  }
}

__global__ void zero_f32(float* __restrict__ p, int n) {
  int i = blockIdx.x * blockDim.x + threadIdx.x;
  if (i < n) p[i] = 0.0f;
}

// ---------------------------------------------------------------------------
// stage 1: QKV projection.  one wave = one 16x16 tile of [S x DH] for one
// (matrix, b, h).  Weights pre-transposed to [H, DH, D] (K-contiguous).
// V is written TRANSPOSED [B,H,DH,S] for K-contiguous PV loads later.
// ---------------------------------------------------------------------------
__global__ void qkv_proj_kernel(const _Float16* __restrict__ xh,
                                const _Float16* __restrict__ Wqt,
                                const _Float16* __restrict__ Wkt,
                                const _Float16* __restrict__ Wvt,
                                const float* __restrict__ bq,
                                const float* __restrict__ bk,
                                const float* __restrict__ bv,
                                _Float16* __restrict__ Qh,
                                _Float16* __restrict__ Kh,
                                _Float16* __restrict__ Vt) {
  int lane = threadIdx.x & 31;
  int wave = blockIdx.x * 8 + (threadIdx.x >> 5);
  int mat  = wave / 16384;
  int rem  = wave % 16384;
  int bh   = rem >> 8;          // b*H + h
  int st   = (rem >> 2) & 63;   // 16-row tile
  int nt   = rem & 3;           // 16-col tile over DH
  int b    = bh >> 4;
  int h    = bh & 15;

  const _Float16* W    = (mat == 0) ? Wqt : (mat == 1) ? Wkt : Wvt;
  const float*    bias = (mat == 0) ? bq  : (mat == 1) ? bk  : bv;

  int c = lane & 15;
  int g = lane >> 4;

  const _Float16* Arow = xh + ((size_t)b * CS + (size_t)st * 16 + c) * CD;
  const _Float16* Bcol = W + ((size_t)h * CDH + (size_t)nt * 16 + c) * CD;

  v8f acc = {};
#pragma unroll 4
  for (int kk = 0; kk < CD; kk += 32) {
    v16h a  = load_afrag(Arow + kk, g);
    v16h bf = load_bfrag(Bcol + kk, g);
    acc = wmma_f16(a, bf, acc);
  }

  float bv_ = bias[h * CDH + nt * 16 + c];
  if (mat == 2) {
    // transposed store: Vt[bh][dh][key]
#pragma unroll
    for (int j = 0; j < 8; ++j) {
      int r = j + 8 * g;
      Vt[((size_t)bh * CDH + (size_t)nt * 16 + c) * CS + (size_t)st * 16 + r] =
          (_Float16)(acc[j] + bv_);
    }
  } else {
    _Float16* Out = (mat == 0) ? Qh : Kh;
#pragma unroll
    for (int j = 0; j < 8; ++j) {
      int r = j + 8 * g;
      Out[((size_t)bh * CS + (size_t)st * 16 + r) * CDH + nt * 16 + c] =
          (_Float16)(acc[j] + bv_);
    }
  }
}

// ---------------------------------------------------------------------------
// stage 2: causal flash attention, heads summed via atomicAdd.
// one wave = one 16-query tile of one (b,h).  V consumed from [B,H,DH,S].
// ---------------------------------------------------------------------------
__global__ void attn_kernel(const _Float16* __restrict__ Qh,
                            const _Float16* __restrict__ Kh,
                            const _Float16* __restrict__ Vt,
                            float* __restrict__ summed) {
  __shared__ _Float16 lds[8][16 * 32];
  int widx = threadIdx.x >> 5;
  int lane = threadIdx.x & 31;
  int wave = blockIdx.x * 8 + widx;
  int qt   = wave & 63;
  int bh   = wave >> 6;
  int b    = bh >> 4;

  int c = lane & 15;
  int g = lane >> 4;

  const _Float16* qrowp = Qh + ((size_t)bh * CS + (size_t)qt * 16 + c) * CDH;
  const _Float16* kbase = Kh + (size_t)bh * CS * CDH;
  const _Float16* vb0 = Vt + ((size_t)bh * CDH + 0 + c) * CS;
  const _Float16* vb1 = Vt + ((size_t)bh * CDH + 16 + c) * CS;
  const _Float16* vb2 = Vt + ((size_t)bh * CDH + 32 + c) * CS;
  const _Float16* vb3 = Vt + ((size_t)bh * CDH + 48 + c) * CS;

  // loop-invariant Q fragments (DH=64 -> two k-steps of 32)
  v16h aq0 = load_afrag(qrowp, g);
  v16h aq1 = load_afrag(qrowp + 32, g);

  float m_run[8], l_run[8];
  v8f O0 = {}, O1 = {}, O2 = {}, O3 = {};
#pragma unroll
  for (int j = 0; j < 8; ++j) { m_run[j] = -3.0e38f; l_run[j] = 0.0f; }

  const float scale = 0.125f;  // 1/sqrt(64)
  int nkb = (qt >> 1) + 1;     // causal 32-key blocks

  for (int kb = 0; kb < nkb; ++kb) {
    int key0 = kb * 32;

    // ---- scores: two 16x16 tiles over this 32-key block -------------------
    const _Float16* krowA = kbase + (size_t)(key0 + c) * CDH;       // keys 0-15
    const _Float16* krowB = kbase + (size_t)(key0 + 16 + c) * CDH;  // keys 16-31
    v16h bka0 = load_bfrag(krowA, g);
    v16h bka1 = load_bfrag(krowA + 32, g);
    v16h bkb0 = load_bfrag(krowB, g);
    v16h bkb1 = load_bfrag(krowB + 32, g);

    v8f Sa = {}, Sb = {};
    Sa = wmma_f16(aq0, bka0, Sa);
    Sa = wmma_f16(aq1, bka1, Sa);
    Sb = wmma_f16(aq0, bkb0, Sb);
    Sb = wmma_f16(aq1, bkb1, Sb);

    // ---- online softmax ---------------------------------------------------
#pragma unroll
    for (int j = 0; j < 8; ++j) {
      int r    = j + 8 * g;
      int qrow = qt * 16 + r;
      float sa = Sa[j] * scale;
      float sb = Sb[j] * scale;
      if (key0 + c > qrow)      sa += NEGMASK;
      if (key0 + 16 + c > qrow) sb += NEGMASK;

      float tm    = rowmax16(fmaxf(sa, sb));
      float m_new = fmaxf(m_run[j], tm);
      float corr  = __expf(m_run[j] - m_new);
      float pa    = __expf(sa - m_new);
      float pb    = __expf(sb - m_new);
      float rs    = rowsum16(pa + pb);
      l_run[j]    = l_run[j] * corr + rs;
      m_run[j]    = m_new;
      O0[j] *= corr; O1[j] *= corr; O2[j] *= corr; O3[j] *= corr;

      lds[widx][r * 32 + c]      = (_Float16)pa;
      lds[widx][r * 32 + 16 + c] = (_Float16)pb;
    }

    // ---- transpose P (D-layout) -> A-layout via per-wave LDS --------------
    v16h ap = load_afrag((const _Float16*)&lds[widx][c * 32], g);

    // ---- O += P @ V  (V is [dh][key]: K-contiguous B loads) ---------------
    v16h bv0 = load_bfrag(vb0 + key0, g);
    v16h bv1 = load_bfrag(vb1 + key0, g);
    v16h bv2 = load_bfrag(vb2 + key0, g);
    v16h bv3 = load_bfrag(vb3 + key0, g);
    O0 = wmma_f16(ap, bv0, O0);
    O1 = wmma_f16(ap, bv1, O1);
    O2 = wmma_f16(ap, bv2, O2);
    O3 = wmma_f16(ap, bv3, O3);
  }

  // ---- normalize and head-sum (reference sums heads) ----------------------
#pragma unroll
  for (int j = 0; j < 8; ++j) {
    int r = j + 8 * g;
    float inv = 1.0f / l_run[j];
    size_t base = ((size_t)b * CS + (size_t)qt * 16 + r) * CDH;
    atomicAdd(&summed[base + 0 + c],  O0[j] * inv);
    atomicAdd(&summed[base + 16 + c], O1[j] * inv);
    atomicAdd(&summed[base + 32 + c], O2[j] * inv);
    atomicAdd(&summed[base + 48 + c], O3[j] * inv);
  }
}

// ---------------------------------------------------------------------------
// stage 3: FF  out = gelu_erf(summed @ Wf + bf).  Wf pre-transposed [D, DH].
// ---------------------------------------------------------------------------
__global__ void ff_gelu_kernel(const float* __restrict__ summed,
                               const _Float16* __restrict__ Wft,
                               const float* __restrict__ bf,
                               float* __restrict__ out) {
  int lane = threadIdx.x & 31;
  int wave = blockIdx.x * 8 + (threadIdx.x >> 5);
  int mt = wave >> 6;   // 16-row tile over B*S
  int nt = wave & 63;   // 16-col tile over D
  int c = lane & 15;
  int g = lane >> 4;

  const float* Arow = summed + ((size_t)mt * 16 + c) * CDH;
  const _Float16* Bcol = Wft + ((size_t)nt * 16 + c) * CDH;

  v16h a0 = cat8(cvt8(Arow + g * 8), cvt8(Arow + 16 + g * 8));
  v16h a1 = cat8(cvt8(Arow + 32 + g * 8), cvt8(Arow + 48 + g * 8));
  v16h b0 = load_bfrag(Bcol, g);
  v16h b1 = load_bfrag(Bcol + 32, g);

  v8f acc = {};
  acc = wmma_f16(a0, b0, acc);
  acc = wmma_f16(a1, b1, acc);

  float bias = bf[nt * 16 + c];
#pragma unroll
  for (int j = 0; j < 8; ++j) {
    int r = j + 8 * g;
    float x = acc[j] + bias;
    float ge = 0.5f * x * (1.0f + erff(x * 0.70710678118654752f));
    out[((size_t)mt * 16 + r) * CD + (size_t)nt * 16 + c] = ge;
  }
}

// ---------------------------------------------------------------------------
// launch
// ---------------------------------------------------------------------------
extern "C" void kernel_launch(void* const* d_in, const int* in_sizes, int n_in,
                              void* d_out, int out_size, void* d_ws,
                              size_t ws_size, hipStream_t stream) {
  const float* x  = (const float*)d_in[0];
  const float* Wq = (const float*)d_in[1];
  const float* bq = (const float*)d_in[2];
  const float* Wk = (const float*)d_in[3];
  const float* bk = (const float*)d_in[4];
  const float* Wv = (const float*)d_in[5];
  const float* bv = (const float*)d_in[6];
  const float* Wf = (const float*)d_in[7];
  const float* bf = (const float*)d_in[8];
  float* out = (float*)d_out;

  const size_t nX  = (size_t)CB * CS * CD;        // 4M
  const size_t nW  = (size_t)CH * CD * CDH;       // 1M each
  const size_t nWf = (size_t)CDH * CD;            // 64K
  const size_t nQ  = (size_t)CB * CH * CS * CDH;  // 4M each
  const size_t nS  = (size_t)CB * CS * CDH;       // 256K floats

  _Float16* xh  = (_Float16*)d_ws;
  _Float16* Wqt = xh + nX;
  _Float16* Wkt = Wqt + nW;
  _Float16* Wvt = Wkt + nW;
  _Float16* Wft = Wvt + nW;
  _Float16* Qh  = Wft + nWf;
  _Float16* Kh  = Qh + nQ;
  _Float16* Vt  = Kh + nQ;
  float* summed = (float*)(Vt + nQ);

  // stage 0: conversions (+ transposes for all B-side operands)
  cvt_f32_to_f16<<<(int)((nX + 255) / 256), 256, 0, stream>>>(x, xh, (int)nX);
  cvt_transpose_f16<<<(int)((nW + 255) / 256), 256, 0, stream>>>(Wq, Wqt, CH,
                                                                 CD, CDH);
  cvt_transpose_f16<<<(int)((nW + 255) / 256), 256, 0, stream>>>(Wk, Wkt, CH,
                                                                 CD, CDH);
  cvt_transpose_f16<<<(int)((nW + 255) / 256), 256, 0, stream>>>(Wv, Wvt, CH,
                                                                 CD, CDH);
  cvt_transpose_f16<<<(int)((nWf + 255) / 256), 256, 0, stream>>>(Wf, Wft, 1,
                                                                  CDH, CD);
  zero_f32<<<(int)((nS + 255) / 256), 256, 0, stream>>>(summed, (int)nS);

  // stage 1: QKV projections
  qkv_proj_kernel<<<6144, 256, 0, stream>>>(xh, Wqt, Wkt, Wvt, bq, bk, bv,
                                            Qh, Kh, Vt);
  // stage 2: causal flash attention + head sum
  attn_kernel<<<512, 256, 0, stream>>>(Qh, Kh, Vt, summed);
  // stage 3: FF + exact GELU
  ff_gelu_kernel<<<2048, 256, 0, stream>>>(summed, Wft, bf, out);
}
